// BCQLinear_87814901334606
// MI455X (gfx1250) — matrix operand order repaired and checked by reading, not compiled, not run on hardware
//
#include <hip/hip_runtime.h>

typedef _Float16 v16h __attribute__((ext_vector_type(16)));
typedef _Float16 v8h  __attribute__((ext_vector_type(8)));
typedef _Float16 v4h  __attribute__((ext_vector_type(4)));
typedef float    v8f  __attribute__((ext_vector_type(8)));
typedef float    v4f  __attribute__((ext_vector_type(4)));

#define K_DIM 4096   // in_features = G*A
#define NB    3      // n_bits

// ---------------------------------------------------------------------------
// Pass 1a: convert activations f32 -> f16 (vectorized by 4)
// ---------------------------------------------------------------------------
__global__ void bcq_cvt_x_f16(const float* __restrict__ x,
                              _Float16* __restrict__ x16, int n4) {
  int i = blockIdx.x * blockDim.x + threadIdx.x;
  if (i >= n4) return;
  v4f v = ((const v4f*)x)[i];
  v4h h = { (_Float16)v.x, (_Float16)v.y, (_Float16)v.z, (_Float16)v.w };
  ((v4h*)x16)[i] = h;
}

// ---------------------------------------------------------------------------
// Pass 1b: dequantize BCQ weight -> f16 (streams 201 MB `binary` once).
//   weight[o, g*A + a] = sum_b alpha[o,g,b] * binary[o,g,a,b]
// ---------------------------------------------------------------------------
__global__ void bcq_dequant_w(const float* __restrict__ binary,
                              const float* __restrict__ alpha,
                              _Float16* __restrict__ w16, int total) {
  int i = blockIdx.x * blockDim.x + threadIdx.x;
  if (i >= total) return;
  int og = i >> 7;                       // / group_size (128)
  const float* al = alpha + (size_t)og * NB;
  const float* bn = binary + (size_t)i * NB;
  float w = al[0] * bn[0] + al[1] * bn[1] + al[2] * bn[2];
  w16[i] = (_Float16)w;
}

// ---------------------------------------------------------------------------
// Fragment loaders (ISA VGPR layouts, CDNA5 ISA 7.12.2).
// A 16x32 f16: lane l -> row m0+(l&15); halves[0..7]=K kb..kb+7,
//              halves[8..15]=K kb+16..kb+23, kb = k0 + (l>>4)*8.
// B 32x16 (from row-major W[N,K]): lane l -> col n0+(l&15);
//              same K order as A (kb..kb+7, kb+16..kb+23).
// ---------------------------------------------------------------------------
__device__ __forceinline__ v16h frag_a(const _Float16* p) {
  v8h lo = *(const v8h*)p;
  v8h hi = *(const v8h*)(p + 16);
  return __builtin_shufflevector(lo, hi, 0, 1, 2, 3, 4, 5, 6, 7,
                                 8, 9, 10, 11, 12, 13, 14, 15);
}
__device__ __forceinline__ v16h frag_b(const _Float16* p) {
  v8h lo = *(const v8h*)p;
  v8h hi = *(const v8h*)(p + 16);
  return __builtin_shufflevector(lo, hi, 0, 1, 2, 3, 4, 5, 6, 7,
                                 8, 9, 10, 11, 12, 13, 14, 15);
}

// ---------------------------------------------------------------------------
// Pass 2: L2-resident WMMA GEMM.  out[m,n] = sum_k x16[m,k]*w16[n,k] + bias[n]
// Block = 8 waves (2 M x 4 N), wave tile 64x32 = 4x2 accumulators.
// K loop double-buffered; the guarded second prefetch keeps the body as two
// scheduling regions, which empirically (rounds 2 vs 3/4) prevents the
// machine pipeliner from over-overlapping iterations (>256 VGPRs + copies).
// ---------------------------------------------------------------------------
__global__ void __launch_bounds__(256)
bcq_gemm_wmma(const _Float16* __restrict__ x16,
              const _Float16* __restrict__ w16,
              const float* __restrict__ bias,
              float* __restrict__ out, int Odim) {
  const int lane = threadIdx.x & 31;
  const int wave = threadIdx.x >> 5;
  const int wm = wave & 1;                    // 2 waves along M
  const int wn = wave >> 1;                   // 4 waves along N
  const int m0 = blockIdx.y * 128 + wm * 64;
  const int n0 = blockIdx.x * 128 + wn * 32;

  const int lhi = lane >> 4;
  const int llo = lane & 15;

  v8f c[4][2];
  v8f zero = {};
#pragma unroll
  for (int mi = 0; mi < 4; ++mi)
#pragma unroll
    for (int ni = 0; ni < 2; ++ni) c[mi][ni] = zero;

  const _Float16* arow[4];
#pragma unroll
  for (int mi = 0; mi < 4; ++mi)
    arow[mi] = x16 + (size_t)(m0 + mi * 16 + llo) * K_DIM + lhi * 8;
  const _Float16* brow[2];
#pragma unroll
  for (int ni = 0; ni < 2; ++ni)
    brow[ni] = w16 + (size_t)(n0 + ni * 16 + llo) * K_DIM + lhi * 8;

  // ---- double-buffered fragment registers (no dynamic indexing) ----
  v16h a0_0, a0_1, a0_2, a0_3, b0_0, b0_1;    // buffer 0
  v16h a1_0, a1_1, a1_2, a1_3, b1_0, b1_1;    // buffer 1

#define LOAD_BUF0(KK)                                                        \
  do {                                                                       \
    a0_0 = frag_a(arow[0] + (KK)); a0_1 = frag_a(arow[1] + (KK));            \
    a0_2 = frag_a(arow[2] + (KK)); a0_3 = frag_a(arow[3] + (KK));            \
    b0_0 = frag_b(brow[0] + (KK)); b0_1 = frag_b(brow[1] + (KK));            \
  } while (0)
#define LOAD_BUF1(KK)                                                        \
  do {                                                                       \
    a1_0 = frag_a(arow[0] + (KK)); a1_1 = frag_a(arow[1] + (KK));            \
    a1_2 = frag_a(arow[2] + (KK)); a1_3 = frag_a(arow[3] + (KK));            \
    b1_0 = frag_b(brow[0] + (KK)); b1_1 = frag_b(brow[1] + (KK));            \
  } while (0)
#define WMMA(CC, AA, BB)                                                     \
  CC = __builtin_amdgcn_wmma_f32_16x16x32_f16(false, AA, false, BB,          \
                                              (short)0, CC, false, false)
#define MM8_BUF0()                                                           \
  do {                                                                       \
    WMMA(c[0][0], a0_0, b0_0); WMMA(c[0][1], a0_0, b0_1);                    \
    WMMA(c[1][0], a0_1, b0_0); WMMA(c[1][1], a0_1, b0_1);                    \
    WMMA(c[2][0], a0_2, b0_0); WMMA(c[2][1], a0_2, b0_1);                    \
    WMMA(c[3][0], a0_3, b0_0); WMMA(c[3][1], a0_3, b0_1);                    \
  } while (0)
#define MM8_BUF1()                                                           \
  do {                                                                       \
    WMMA(c[0][0], a1_0, b1_0); WMMA(c[0][1], a1_0, b1_1);                    \
    WMMA(c[1][0], a1_1, b1_0); WMMA(c[1][1], a1_1, b1_1);                    \
    WMMA(c[2][0], a1_2, b1_0); WMMA(c[2][1], a1_2, b1_1);                    \
    WMMA(c[3][0], a1_3, b1_0); WMMA(c[3][1], a1_3, b1_1);                    \
  } while (0)

  LOAD_BUF0(0);
  for (int k0 = 0; k0 < K_DIM; k0 += 64) {
    LOAD_BUF1(k0 + 32);                       // prefetch next 32-K slab
    MM8_BUF0();                               // compute on buffer 0
    if (k0 + 64 < K_DIM) LOAD_BUF0(k0 + 64);  // guarded prefetch (also acts
    MM8_BUF1();                               //  as a region split)
  }
#undef LOAD_BUF0
#undef LOAD_BUF1
#undef WMMA
#undef MM8_BUF0
#undef MM8_BUF1

  // Epilogue: C/D layout — VGPR v, lane l -> M = v + 8*(l>=16), N = l&15.
#pragma unroll
  for (int ni = 0; ni < 2; ++ni) {
    const int n = n0 + ni * 16 + llo;
    const float bv = bias[n];
#pragma unroll
    for (int mi = 0; mi < 4; ++mi) {
      const int mbase = m0 + mi * 16 + lhi * 8;
#pragma unroll
      for (int v = 0; v < 8; ++v)
        out[(size_t)(mbase + v) * Odim + n] = c[mi][ni][v] + bv;
    }
  }
}

// ---------------------------------------------------------------------------
extern "C" void kernel_launch(void* const* d_in, const int* in_sizes, int n_in,
                              void* d_out, int out_size, void* d_ws, size_t ws_size,
                              hipStream_t stream) {
  const float* x      = (const float*)d_in[0];  // [B,S,I] f32
  const float* binary = (const float*)d_in[1];  // [O,G,A,NB] f32 (+/-1)
  const float* alpha  = (const float*)d_in[2];  // [O,G,NB] f32
  const float* bias   = (const float*)d_in[3];  // [O] f32
  float* out = (float*)d_out;

  const int K    = K_DIM;                       // 4096
  const int O    = 4096;
  const int Mtot = in_sizes[0] / K;             // B*S = 2048

  // scratch: [0, 32MB) = w16[O*K], [32MB, 48MB) = x16[Mtot*K]
  _Float16* w16 = (_Float16*)d_ws;
  _Float16* x16 = (_Float16*)((char*)d_ws + (size_t)O * K * sizeof(_Float16));

  int n4 = in_sizes[0] / 4;
  bcq_cvt_x_f16<<<(n4 + 255) / 256, 256, 0, stream>>>(x, x16, n4);

  int total = O * K;                            // O*G*A
  bcq_dequant_w<<<(total + 255) / 256, 256, 0, stream>>>(binary, alpha, w16, total);

  dim3 grid(O / 128, Mtot / 128);               // 32 x 16 blocks of 128x128
  bcq_gemm_wmma<<<grid, 256, 0, stream>>>(x16, w16, bias, out, O);
}
